// FlowNet3DiscriminatorHCat_1872605741577
// MI455X (gfx1250) — matrix operand order repaired and checked by reading, not compiled, not run on hardware
//
#include <hip/hip_runtime.h>

// ---------------------------------------------------------------------------
// FlowNet3D discriminator forward for MI455X (gfx1250, wave32, WMMA).
// Conv stacks (the FLOP-dominant part) run as f16-in / f32-acc GEMMs on
// v_wmma_f32_16x16x32_f16.  Activations are stored K-major [Kpad][Ntot] so
// each K row of a wave's B tile (64 N values across its lanes) is one contiguous
// 128-byte run; the shared A panel (16 x Kpad f16) is staged per-block into
// LDS via global_load_async_to_lds_b128 (+ s_wait_asynccnt) and read back as
// ds_load_b128 fragments.  Each wave computes a 16x64 tile (4 accumulators,
// A-fragment reuse) for ~13 FLOP/byte of fragment traffic.
// ---------------------------------------------------------------------------

typedef _Float16 h16;
typedef __attribute__((ext_vector_type(16))) _Float16 v16h;
typedef __attribute__((ext_vector_type(8)))  float    v8f;

#define BB 8  // batch

static inline int ceil32i(int x) { return (x + 31) & ~31; }
static inline int cdiv(int a, int b) { return (a + b - 1) / b; }

// ------------------------------ FPS ----------------------------------------
// One workgroup per batch. dist[] lives in LDS (Nin <= 8192 -> 32KB).
__global__ void fps_kernel(const float* __restrict__ xyz, int Nin, int npoint,
                           int* __restrict__ idxOut) {
  int b = blockIdx.x;
  const float* px = xyz + (size_t)b * 3 * Nin;  // [3][Nin]
  extern __shared__ float smem[];
  float* dist = smem;                 // Nin
  float* rv   = smem + Nin;           // 256
  int*   ri   = (int*)(rv + 256);     // 256
  int t = threadIdx.x;
  for (int i = t; i < Nin; i += 256) dist[i] = 1e10f;
  __syncthreads();
  int far = 0;
  for (int it = 0; it < npoint; ++it) {
    if (t == 0) idxOut[(size_t)b * npoint + it] = far;
    float cx = px[far], cy = px[Nin + far], cz = px[2 * Nin + far];
    float best = -1.0f; int bi = 0;
    for (int i = t; i < Nin; i += 256) {
      float dx = px[i] - cx, dy = px[Nin + i] - cy, dz = px[2 * Nin + i] - cz;
      float d  = dx * dx + dy * dy + dz * dz;
      float dd = fminf(dist[i], d);
      dist[i] = dd;
      if (dd > best) { best = dd; bi = i; }
    }
    rv[t] = best; ri[t] = bi;
    __syncthreads();
    for (int s = 128; s > 0; s >>= 1) {
      if (t < s) {
        if (rv[t + s] > rv[t] || (rv[t + s] == rv[t] && ri[t + s] < ri[t])) {
          rv[t] = rv[t + s]; ri[t] = ri[t + s];
        }
      }
      __syncthreads();
    }
    far = ri[0];
    __syncthreads();
  }
}

// new_xyz[b][c][s] = xyz[b][c][idx[b][s]]
__global__ void gather_coords_kernel(const float* __restrict__ xyz,
                                     const int* __restrict__ idx,
                                     float* __restrict__ out, int Nin, int S) {
  int gid = blockIdx.x * blockDim.x + threadIdx.x;
  int total = BB * 3 * S;
  if (gid >= total) return;
  int b = gid / (3 * S);
  int r = gid % (3 * S);
  int c = r / S, s = r % S;
  out[gid] = xyz[((size_t)b * 3 + c) * Nin + idx[(size_t)b * S + s]];
}

// --------------------------- ball query ------------------------------------
__global__ void ball_query_kernel(const float* __restrict__ pts,
                                  const float* __restrict__ ctr,
                                  int* __restrict__ grp,
                                  int Nin, int S, int K, float r2) {
  int gid = blockIdx.x * blockDim.x + threadIdx.x;
  if (gid >= BB * S) return;
  int b = gid / S, s = gid % S;
  const float* px = pts + (size_t)b * 3 * Nin;
  float cx = ctr[((size_t)b * 3 + 0) * S + s];
  float cy = ctr[((size_t)b * 3 + 1) * S + s];
  float cz = ctr[((size_t)b * 3 + 2) * S + s];
  int* g = grp + (size_t)gid * K;
  int cnt = 0;
  for (int j = 0; j < Nin && cnt < K; ++j) {
    float dx = px[j] - cx, dy = px[Nin + j] - cy, dz = px[2 * Nin + j] - cz;
    if (dx * dx + dy * dy + dz * dz <= r2) g[cnt++] = j;
  }
  int fill = cnt ? g[0] : (Nin - 1);
  for (int k = cnt; k < K; ++k) g[k] = fill;
}

// ------------------------ grouping / concat ---------------------------------
// Builds the GEMM input X (f16, K-major [Kpad][Ntot]); col = (b*S+s)*K + k.
__global__ void sa_group_kernel(const float* __restrict__ xyz,
                                const float* __restrict__ feats,
                                const float* __restrict__ ctr,
                                const int* __restrict__ grp,
                                h16* __restrict__ X,
                                int Nin, int S, int K, int D, int Kpad, int Ntot) {
  int col = blockIdx.x * blockDim.x + threadIdx.x;
  if (col >= Ntot) return;
  int k = col % K;
  int bs = col / K;
  int s = bs % S, b = bs / S;
  int j = grp[(size_t)(b * S + s) * K + k];
  const float* px = xyz + (size_t)b * 3 * Nin;
  for (int c = 0; c < 3; ++c)
    X[(size_t)c * Ntot + col] =
        (h16)(px[(size_t)c * Nin + j] - ctr[((size_t)b * 3 + c) * S + s]);
  const float* pf = feats + (size_t)b * D * Nin;
  for (int c = 0; c < D; ++c)
    X[(size_t)(3 + c) * Ntot + col] = (h16)pf[(size_t)c * Nin + j];
  for (int c = 3 + D; c < Kpad; ++c) X[(size_t)c * Ntot + col] = (h16)0.0f;
}

// flow-embedding concat: [pos_diff(3), grouped f2(D2), repeated f1(D1)]
__global__ void fe_group_kernel(const float* __restrict__ p2,
                                const float* __restrict__ p1,
                                const float* __restrict__ f2,
                                const float* __restrict__ f1,
                                const int* __restrict__ grp,
                                h16* __restrict__ X,
                                int N2, int S, int K, int D2, int D1,
                                int Kpad, int Ntot) {
  int col = blockIdx.x * blockDim.x + threadIdx.x;
  if (col >= Ntot) return;
  int k = col % K;
  int bs = col / K;
  int s = bs % S, b = bs / S;
  int j = grp[(size_t)(b * S + s) * K + k];
  const float* px = p2 + (size_t)b * 3 * N2;
  for (int c = 0; c < 3; ++c)
    X[(size_t)c * Ntot + col] =
        (h16)(px[(size_t)c * N2 + j] - p1[((size_t)b * 3 + c) * S + s]);
  const float* pf2 = f2 + (size_t)b * D2 * N2;
  for (int c = 0; c < D2; ++c)
    X[(size_t)(3 + c) * Ntot + col] = (h16)pf2[(size_t)c * N2 + j];
  const float* pf1 = f1 + (size_t)b * D1 * S;
  for (int c = 0; c < D1; ++c)
    X[(size_t)(3 + D2 + c) * Ntot + col] = (h16)pf1[(size_t)c * S + s];
  for (int c = 3 + D2 + D1; c < Kpad; ++c)
    X[(size_t)c * Ntot + col] = (h16)0.0f;
}

// pack conv weight [Cout][Cin] f32 -> [Cout][Kpad] f16 (zero padded)
__global__ void pack_weight_kernel(const float* __restrict__ W,
                                   h16* __restrict__ Wh,
                                   int Cout, int Cin, int Kpad) {
  int gid = blockIdx.x * blockDim.x + threadIdx.x;
  if (gid >= Cout * Kpad) return;
  int o = gid / Kpad, c = gid % Kpad;
  Wh[gid] = (h16)((c < Cin) ? W[(size_t)o * Cin + c] : 0.0f);
}

// ------------------------------ WMMA GEMM -----------------------------------
// Y[M][Nt] f32 = A[M][Kpad] f16 * B[Kpad][Nt] f16 + bias[M].
// Block = 8 waves, block-uniform tileM.  A panel async-staged to LDS once per
// block; each wave computes a 16x64 tile (4x v_wmma_f32_16x16x32_f16 per
// 32-deep K chunk, A fragment reused).
union Frag16 { v16h v; float4 q[2]; };

#define MAX_KPAD 512

__global__ __launch_bounds__(256)
void wmma_gemm_kernel(const h16* __restrict__ A, const h16* __restrict__ Bm,
                      const float* __restrict__ bias, float* __restrict__ Y,
                      int M, int Kpad, int Nt) {
  __shared__ h16 shA[16 * MAX_KPAD];
  int nbN   = (Nt >> 6) >> 3;               // groups of 8 waves x 64 cols
  int tileM = blockIdx.x / nbN;             // block-uniform M tile
  int nb    = blockIdx.x % nbN;
  int waveN = nb * 8 + (threadIdx.x >> 5);  // this wave's 64-col tile
  int lane  = threadIdx.x & 31;

  // ---- async-stage A tile (16 rows x Kpad f16, rows contiguous) into LDS ----
  {
    const h16* gA = A + (size_t)tileM * 16 * Kpad;
    unsigned bytes = 16u * (unsigned)Kpad * 2u;
    for (unsigned off = (unsigned)threadIdx.x * 16u; off < bytes;
         off += 256u * 16u) {
      unsigned long long ga = (unsigned long long)(uintptr_t)gA + off;
      unsigned la = (unsigned)(uintptr_t)shA + off;  // LDS aperture: low 32b
      asm volatile("global_load_async_to_lds_b128 %0, %1, off"
                   :: "v"(la), "v"(ga) : "memory");
    }
    asm volatile("s_wait_asynccnt 0" ::: "memory");
  }
  __syncthreads();

  int g = lane >> 4;      // half-select
  int l = lane & 15;
  const h16* arow = shA + (size_t)l * Kpad;  // A row M = l (both halves)
  v8f acc0 = {}, acc1 = {}, acc2 = {}, acc3 = {};
  for (int kk = 0; kk < Kpad; kk += 32) {
    Frag16 a, b0, b1, b2, b3;
    // A: elems 0..7 = K kb..kb+7, elems 8..15 = K kb+16..kb+23, kb = kk+8g
    const h16* ap = arow + kk + g * 8;
    a.q[0] = *(const float4*)(ap);           // ds_load_b128
    a.q[1] = *(const float4*)(ap + 16);
    // B: lane holds col N = l of each 16-col tile; elems = K same order as A
    const h16* bp = Bm + (size_t)(kk + g * 8) * Nt + waveN * 64 + l;
    for (int s = 0; s < 16; ++s) {
      const h16* bs = bp + (size_t)(s < 8 ? s : s + 8) * Nt;
      b0.v[s] = bs[0]; b1.v[s] = bs[16]; b2.v[s] = bs[32]; b3.v[s] = bs[48];
    }
    if (kk + 32 < Kpad)
      __builtin_prefetch(bp + (size_t)32 * Nt, 0, 0);  // global_prefetch_b8
    acc0 = __builtin_amdgcn_wmma_f32_16x16x32_f16(false, a.v, false, b0.v,
                                                  (short)0, acc0, false, false);
    acc1 = __builtin_amdgcn_wmma_f32_16x16x32_f16(false, a.v, false, b1.v,
                                                  (short)0, acc1, false, false);
    acc2 = __builtin_amdgcn_wmma_f32_16x16x32_f16(false, a.v, false, b2.v,
                                                  (short)0, acc2, false, false);
    acc3 = __builtin_amdgcn_wmma_f32_16x16x32_f16(false, a.v, false, b3.v,
                                                  (short)0, acc3, false, false);
  }
  // D: lane<16 -> M=r, lane>=16 -> M=r+8; N = lane&15 within each 16-col tile
#pragma unroll
  for (int r = 0; r < 8; ++r) {
    int m = tileM * 16 + r + g * 8;
    float bsum = bias[m];
    float* yr = Y + (size_t)m * Nt + (size_t)waveN * 64 + l;
    yr[0]  = acc0[r] + bsum;
    yr[16] = acc1[r] + bsum;
    yr[32] = acc2[r] + bsum;
    yr[48] = acc3[r] + bsum;
  }
}

// -------------------------- BatchNorm (train) -------------------------------
__global__ void bn_stats_kernel(const float* __restrict__ Y,
                                float* __restrict__ mean, float* __restrict__ istd,
                                int Nt) {
  int m = blockIdx.x, t = threadIdx.x;
  const float* row = Y + (size_t)m * Nt;
  float s1 = 0.0f, s2 = 0.0f;
  for (int i = t; i < Nt; i += 256) { float v = row[i]; s1 += v; s2 += v * v; }
  __shared__ float a1[256], a2[256];
  a1[t] = s1; a2[t] = s2;
  __syncthreads();
  for (int s = 128; s > 0; s >>= 1) {
    if (t < s) { a1[t] += a1[t + s]; a2[t] += a2[t + s]; }
    __syncthreads();
  }
  if (t == 0) {
    float mu = a1[0] / (float)Nt;
    float var = a2[0] / (float)Nt - mu * mu;
    mean[m] = mu;
    istd[m] = rsqrtf(var + 1e-5f);
  }
}

__global__ void bn_relu_kernel(const float* __restrict__ Y,
                               const float* __restrict__ mean,
                               const float* __restrict__ istd,
                               const float* __restrict__ gamma,
                               const float* __restrict__ beta,
                               h16* __restrict__ outH, float* __restrict__ outF,
                               int C, int Nt) {
  size_t gid = (size_t)blockIdx.x * blockDim.x + threadIdx.x;
  size_t total = (size_t)C * Nt;
  if (gid >= total) return;
  int m = (int)(gid / Nt);
  float v = (Y[gid] - mean[m]) * istd[m] * gamma[m] + beta[m];
  v = fmaxf(v, 0.0f);
  if (outH) outH[gid] = (h16)v;
  if (outF) outF[gid] = v;
}

// ------------------------------- pooling ------------------------------------
// act [C][B*S*K] -> feats [B][C][S], max over k
__global__ void maxpool_k_kernel(const float* __restrict__ act,
                                 float* __restrict__ feats,
                                 int C, int S, int K) {
  int gid = blockIdx.x * blockDim.x + threadIdx.x;
  if (gid >= BB * C * S) return;
  int s = gid % S;
  int c = (gid / S) % C;
  int b = gid / (S * C);
  size_t Nt = (size_t)BB * S * K;
  const float* p = act + (size_t)c * Nt + (size_t)(b * S + s) * K;
  float mx = -3.0e38f;
  for (int k = 0; k < K; ++k) mx = fmaxf(mx, p[k]);
  feats[((size_t)b * C + c) * S + s] = mx;
}

// feats [B][C][S] -> out[b*ld + c], max or mean over S
__global__ void pool_s_kernel(const float* __restrict__ f, float* __restrict__ out,
                              int C, int S, int ld, int domean) {
  int gid = blockIdx.x * blockDim.x + threadIdx.x;
  if (gid >= BB * C) return;
  int c = gid % C, b = gid / C;
  const float* p = f + ((size_t)b * C + c) * S;
  if (domean) {
    float s = 0.0f;
    for (int i = 0; i < S; ++i) s += p[i];
    out[(size_t)b * ld + c] = s / (float)S;
  } else {
    float m = -3.0e38f;
    for (int i = 0; i < S; ++i) m = fmaxf(m, p[i]);
    out[(size_t)b * ld + c] = m;
  }
}

__global__ void copy2d_kernel(const float* __restrict__ src, int lds,
                              float* __restrict__ dst, int ldd,
                              int rows, int cols) {
  int gid = blockIdx.x * blockDim.x + threadIdx.x;
  if (gid >= rows * cols) return;
  int r = gid / cols, c = gid % cols;
  dst[(size_t)r * ldd + c] = src[(size_t)r * lds + c];
}

// ----------------------------- FC head --------------------------------------
__global__ void fc_kernel(const float* __restrict__ X, int din,
                          const float* __restrict__ W, const float* __restrict__ bias,
                          float* __restrict__ Y, int ldy, int dout) {
  int gid = blockIdx.x * blockDim.x + threadIdx.x;
  if (gid >= BB * dout) return;
  int o = gid % dout, i = gid / dout;
  const float* x = X + (size_t)i * din;
  const float* w = W + (size_t)o * din;
  float s = bias[o];
  for (int c = 0; c < din; ++c) s += x[c] * w[c];
  Y[(size_t)i * ldy + o] = s;
}

__global__ void bn8_relu_kernel(float* __restrict__ Y, int ldy,
                                const float* __restrict__ g,
                                const float* __restrict__ be, int dout) {
  int o = blockIdx.x * blockDim.x + threadIdx.x;
  if (o >= dout) return;
  float s1 = 0.0f, s2 = 0.0f;
  for (int i = 0; i < BB; ++i) { float v = Y[(size_t)i * ldy + o]; s1 += v; s2 += v * v; }
  float mu = s1 * (1.0f / BB);
  float var = s2 * (1.0f / BB) - mu * mu;
  float is = rsqrtf(var + 1e-5f);
  for (int i = 0; i < BB; ++i) {
    float v = (Y[(size_t)i * ldy + o] - mu) * is * g[o] + be[o];
    Y[(size_t)i * ldy + o] = fmaxf(v, 0.0f);
  }
}

// ---------------------------------------------------------------------------
extern "C" void kernel_launch(void* const* d_in, const int* in_sizes, int n_in,
                              void* d_out, int out_size, void* d_ws, size_t ws_size,
                              hipStream_t stream) {
  (void)in_sizes; (void)n_in; (void)out_size; (void)ws_size;
  const float* pc1 = (const float*)d_in[0];
  const float* pc2 = (const float*)d_in[1];
  auto P = [&](int i) { return (const float*)d_in[i]; };

  // param base indices (dict order: sa1..sa4, fe, fc1, fc2, fcf, fc3, fc4)
  const int SA1 = 2, SA2 = 14, SA3 = 26, SA4 = 38, FE = 50;
  const int FC1 = 62, FC2 = 70, FCF = 78, FC3 = 86, FC4 = 94;
  const int ch_sa1[4] = {6, 32, 32, 64};
  const int ch_sa2[4] = {67, 64, 64, 128};
  const int ch_sa3[4] = {131, 128, 128, 256};
  const int ch_sa4[4] = {259, 256, 256, 512};
  const int ch_fe[4]  = {259, 128, 128, 128};

  // bump allocator over d_ws
  char* wp = (char*)d_ws;
  auto alloc = [&](size_t bytes) -> void* {
    void* p = (void*)wp;
    wp += (bytes + 255) & ~(size_t)255;
    return p;
  };

  // persistent tensors  ([B,3,S] coords, [B,C,S] feats)
  float* l1_pc1 = (float*)alloc((size_t)BB * 3 * 1024 * 4);
  float* l1_f1  = (float*)alloc((size_t)BB * 64 * 1024 * 4);
  float* l1_pc2 = (float*)alloc((size_t)BB * 3 * 1024 * 4);
  float* l1_f2  = (float*)alloc((size_t)BB * 64 * 1024 * 4);
  float* l2_pc1 = (float*)alloc((size_t)BB * 3 * 256 * 4);
  float* l2_f1  = (float*)alloc((size_t)BB * 128 * 256 * 4);
  float* l2_pc2 = (float*)alloc((size_t)BB * 3 * 256 * 4);
  float* l2_f2  = (float*)alloc((size_t)BB * 128 * 256 * 4);
  float* flow   = (float*)alloc((size_t)BB * 128 * 256 * 4);
  float* l3_pc  = (float*)alloc((size_t)BB * 3 * 64 * 4);
  float* l3_f   = (float*)alloc((size_t)BB * 256 * 64 * 4);
  float* l4_pc  = (float*)alloc((size_t)BB * 3 * 16 * 4);
  float* l4_f   = (float*)alloc((size_t)BB * 512 * 16 * 4);
  // FC staging
  float* m_l1f1 = (float*)alloc((size_t)BB * 64 * 4);
  float* m_l1f2 = (float*)alloc((size_t)BB * 64 * 4);
  float* m_l2f1 = (float*)alloc((size_t)BB * 128 * 4);
  float* m_l2f2 = (float*)alloc((size_t)BB * 128 * 4);
  float* m_flow = (float*)alloc((size_t)BB * 128 * 4);
  float* m_l3   = (float*)alloc((size_t)BB * 256 * 4);
  float* m_l4   = (float*)alloc((size_t)BB * 512 * 4);
  float* f1buf  = (float*)alloc((size_t)BB * 128 * 4);
  float* f2in   = (float*)alloc((size_t)BB * 256 * 4);
  float* f2buf  = (float*)alloc((size_t)BB * 512 * 4);
  float* ffin   = (float*)alloc((size_t)BB * 640 * 4);
  float* ffbuf  = (float*)alloc((size_t)BB * 1024 * 4);
  float* f3in   = (float*)alloc((size_t)BB * 1280 * 4);
  float* f3buf  = (float*)alloc((size_t)BB * 2048 * 4);
  float* outin  = (float*)alloc((size_t)BB * 2560 * 4);
  char* stageBase = wp;  // everything after this is per-stage scratch

  // ---- conv stack: 3x (pack W; WMMA GEMM; BN stats; BN+ReLU) ----
  auto run_conv_stack = [&](int pbase, const int ch[4], int Ntot, h16* X0,
                            float* actOut) {
    int maxMid = ch[1] > ch[2] ? ch[1] : ch[2];
    int maxC = maxMid > ch[3] ? maxMid : ch[3];
    float* Y  = (float*)alloc((size_t)maxC * Ntot * 4);
    h16* Xa   = (h16*)alloc((size_t)maxMid * Ntot * 2);
    h16* Xb   = (h16*)alloc((size_t)maxMid * Ntot * 2);
    h16* Xin = X0;
    int Cin = ch[0];
    for (int L = 0; L < 3; ++L) {
      int Kpad = ceil32i(Cin);
      int Cout = ch[L + 1];
      const float* W  = P(pbase + 4 * L + 0);
      const float* bb = P(pbase + 4 * L + 1);
      const float* gm = P(pbase + 4 * L + 2);
      const float* bt = P(pbase + 4 * L + 3);
      h16* Wh = (h16*)alloc((size_t)Cout * Kpad * 2);
      pack_weight_kernel<<<cdiv(Cout * Kpad, 256), 256, 0, stream>>>(
          W, Wh, Cout, Cin, Kpad);
      // grid: tilesM blocks per 8-wave group of 64-col tiles (Ntot/64 % 8 == 0
      // for every layer shape in this net)
      int grid = (Cout / 16) * ((Ntot / 64) / 8);
      wmma_gemm_kernel<<<grid, 256, 0, stream>>>(Wh, Xin, bb, Y, Cout, Kpad, Ntot);
      float* mean = (float*)alloc((size_t)Cout * 4);
      float* istd = (float*)alloc((size_t)Cout * 4);
      bn_stats_kernel<<<Cout, 256, 0, stream>>>(Y, mean, istd, Ntot);
      h16* Xn = (L < 2) ? ((L == 0) ? Xa : Xb) : nullptr;
      bn_relu_kernel<<<cdiv(Cout * Ntot, 256), 256, 0, stream>>>(
          Y, mean, istd, gm, bt, Xn, (L == 2) ? actOut : nullptr, Cout, Ntot);
      Xin = Xn;
      Cin = Cout;
    }
  };

  // ---- set abstraction ----
  auto run_sa = [&](const float* xyz, int Nin, const float* feats, int D,
                    int S, int K, float radius, int pbase, const int ch[4],
                    float* new_xyz, float* featsOut) {
    char* save = wp;
    int* fpsIdx = (int*)alloc((size_t)BB * S * 4);
    fps_kernel<<<BB, 256, (size_t)Nin * 4 + 2048, stream>>>(xyz, Nin, S, fpsIdx);
    gather_coords_kernel<<<cdiv(BB * 3 * S, 256), 256, 0, stream>>>(
        xyz, fpsIdx, new_xyz, Nin, S);
    int* grp = (int*)alloc((size_t)BB * S * K * 4);
    ball_query_kernel<<<cdiv(BB * S, 128), 128, 0, stream>>>(
        xyz, new_xyz, grp, Nin, S, K, radius * radius);
    int Cin = 3 + D;
    int Kpad0 = ceil32i(Cin);
    int Ntot = BB * S * K;
    h16* X = (h16*)alloc((size_t)Kpad0 * Ntot * 2);
    sa_group_kernel<<<cdiv(Ntot, 256), 256, 0, stream>>>(
        xyz, feats, new_xyz, grp, X, Nin, S, K, D, Kpad0, Ntot);
    float* act = (float*)alloc((size_t)ch[3] * Ntot * 4);
    run_conv_stack(pbase, ch, Ntot, X, act);
    maxpool_k_kernel<<<cdiv(BB * ch[3] * S, 256), 256, 0, stream>>>(
        act, featsOut, ch[3], S, K);
    wp = save;
  };

  // ---- stages ----
  wp = stageBase; run_sa(pc1, 8192, pc1, 3, 1024, 16, 0.5f, SA1, ch_sa1, l1_pc1, l1_f1);
  wp = stageBase; run_sa(l1_pc1, 1024, l1_f1, 64, 256, 16, 1.0f, SA2, ch_sa2, l2_pc1, l2_f1);
  wp = stageBase; run_sa(pc2, 8192, pc2, 3, 1024, 16, 0.5f, SA1, ch_sa1, l1_pc2, l1_f2);
  wp = stageBase; run_sa(l1_pc2, 1024, l1_f2, 64, 256, 16, 1.0f, SA2, ch_sa2, l2_pc2, l2_f2);

  // flow embedding: group p2 around p1, concat [diff, f2[grp], f1 repeat]
  {
    wp = stageBase;
    int S = 256, K = 64, N2 = 256;
    int* grp = (int*)alloc((size_t)BB * S * K * 4);
    ball_query_kernel<<<cdiv(BB * S, 128), 128, 0, stream>>>(
        l2_pc2, l2_pc1, grp, N2, S, K, 100.0f);
    int Cin = 259, Kpad = ceil32i(Cin), Ntot = BB * S * K;
    h16* X = (h16*)alloc((size_t)Kpad * Ntot * 2);
    fe_group_kernel<<<cdiv(Ntot, 256), 256, 0, stream>>>(
        l2_pc2, l2_pc1, l2_f2, l2_f1, grp, X, N2, S, K, 128, 128, Kpad, Ntot);
    float* act = (float*)alloc((size_t)128 * Ntot * 4);
    run_conv_stack(FE, ch_fe, Ntot, X, act);
    maxpool_k_kernel<<<cdiv(BB * 128 * S, 256), 256, 0, stream>>>(
        act, flow, 128, S, K);
  }

  wp = stageBase; run_sa(l2_pc2, 256, flow, 128, 64, 8, 2.0f, SA3, ch_sa3, l3_pc, l3_f);
  wp = stageBase; run_sa(l3_pc, 64, l3_f, 256, 16, 8, 4.0f, SA4, ch_sa4, l4_pc, l4_f);

  // ---- global max pools feeding the FC heads ----
  wp = stageBase;
  pool_s_kernel<<<cdiv(BB * 64, 256), 256, 0, stream>>>(l1_f1, m_l1f1, 64, 1024, 64, 0);
  pool_s_kernel<<<cdiv(BB * 64, 256), 256, 0, stream>>>(l1_f2, m_l1f2, 64, 1024, 64, 0);
  pool_s_kernel<<<cdiv(BB * 128, 256), 256, 0, stream>>>(l2_f1, m_l2f1, 128, 256, 128, 0);
  pool_s_kernel<<<cdiv(BB * 128, 256), 256, 0, stream>>>(l2_f2, m_l2f2, 128, 256, 128, 0);
  pool_s_kernel<<<cdiv(BB * 128, 256), 256, 0, stream>>>(flow, m_flow, 128, 256, 128, 0);
  pool_s_kernel<<<cdiv(BB * 256, 256), 256, 0, stream>>>(l3_f, m_l3, 256, 64, 256, 0);
  pool_s_kernel<<<cdiv(BB * 512, 256), 256, 0, stream>>>(l4_f, m_l4, 512, 16, 512, 0);

  // ---- FC tower ----
  auto run_fc2l = [&](int pbase, const float* X, int d0, int d1, int d2,
                      float* out, int ldOut, bool lastPlain) {
    float* t = (float*)alloc((size_t)BB * d1 * 4);
    fc_kernel<<<cdiv(BB * d1, 256), 256, 0, stream>>>(X, d0, P(pbase), P(pbase + 1), t, d1, d1);
    bn8_relu_kernel<<<cdiv(d1, 256), 256, 0, stream>>>(t, d1, P(pbase + 2), P(pbase + 3), d1);
    fc_kernel<<<cdiv(BB * d2, 256), 256, 0, stream>>>(t, d1, P(pbase + 4), P(pbase + 5), out, ldOut, d2);
    if (!lastPlain)
      bn8_relu_kernel<<<cdiv(d2, 256), 256, 0, stream>>>(out, ldOut, P(pbase + 6), P(pbase + 7), d2);
  };
  auto cat = [&](const float* src, int cols, float* dst, int ldd, int colOff) {
    copy2d_kernel<<<cdiv(BB * cols, 256), 256, 0, stream>>>(src, cols, dst + colOff, ldd, BB, cols);
  };

  // f1 = [fc1(max l1_f1), fc1(max l1_f2)]
  run_fc2l(FC1, m_l1f1, 64, 64, 64, f1buf, 128, false);
  run_fc2l(FC1, m_l1f2, 64, 64, 64, f1buf + 64, 128, false);
  // f2 = [fc2([max l2_f1, f1]), fc2([max l2_f2, f1])]
  cat(m_l2f1, 128, f2in, 256, 0); cat(f1buf, 128, f2in, 256, 128);
  run_fc2l(FC2, f2in, 256, 256, 256, f2buf, 512, false);
  cat(m_l2f2, 128, f2in, 256, 0); cat(f1buf, 128, f2in, 256, 128);
  run_fc2l(FC2, f2in, 256, 256, 256, f2buf + 256, 512, false);
  // ff = fcf([max flow, f2])
  cat(m_flow, 128, ffin, 640, 0); cat(f2buf, 512, ffin, 640, 128);
  run_fc2l(FCF, ffin, 640, 640, 1024, ffbuf, 1024, false);
  // f3 = fc3([max l3_f, ff])
  cat(m_l3, 256, f3in, 1280, 0); cat(ffbuf, 1024, f3in, 1280, 256);
  run_fc2l(FC3, f3in, 1280, 1280, 2048, f3buf, 2048, false);
  // out = fc4([max l4_f, f3])  (last layer plain) -> directly into d_out
  cat(m_l4, 512, outin, 2560, 0); cat(f3buf, 2048, outin, 2560, 512);
  float* dout = (float*)d_out;
  run_fc2l(FC4, outin, 2560, 2560, 4096, dout, 4096, true);

  // ---- hidden outputs, flat after `out` in return order ----
  size_t off = (size_t)BB * 4096;
  copy2d_kernel<<<cdiv(BB * 2048, 256), 256, 0, stream>>>(f3buf, 2048, dout + off, 2048, BB, 2048);
  off += (size_t)BB * 2048;
  copy2d_kernel<<<cdiv(BB * 1024, 256), 256, 0, stream>>>(ffbuf, 1024, dout + off, 1024, BB, 1024);
  off += (size_t)BB * 1024;
  copy2d_kernel<<<cdiv(BB * 512, 256), 256, 0, stream>>>(f2buf, 512, dout + off, 512, BB, 512);
  off += (size_t)BB * 512;
  copy2d_kernel<<<cdiv(BB * 128, 256), 256, 0, stream>>>(f1buf, 128, dout + off, 128, BB, 128);
  off += (size_t)BB * 128;
  pool_s_kernel<<<cdiv(BB * 512, 256), 256, 0, stream>>>(l4_f, dout + off, 512, 16, 512, 1);
  off += (size_t)BB * 512;
  pool_s_kernel<<<cdiv(BB * 256, 256), 256, 0, stream>>>(l3_f, dout + off, 256, 64, 256, 1);
  off += (size_t)BB * 256;
  pool_s_kernel<<<cdiv(BB * 128, 256), 256, 0, stream>>>(flow, dout + off, 128, 256, 128, 1);
  off += (size_t)BB * 128;
  pool_s_kernel<<<cdiv(BB * 128, 256), 256, 0, stream>>>(l2_f2, dout + off, 128, 256, 128, 1);
  off += (size_t)BB * 128;
  pool_s_kernel<<<cdiv(BB * 64, 256), 256, 0, stream>>>(l1_f2, dout + off, 64, 1024, 64, 1);
  off += (size_t)BB * 64;
  pool_s_kernel<<<cdiv(BB * 128, 256), 256, 0, stream>>>(l2_f1, dout + off, 128, 256, 128, 1);
  off += (size_t)BB * 128;
  pool_s_kernel<<<cdiv(BB * 64, 256), 256, 0, stream>>>(l1_f1, dout + off, 64, 1024, 64, 1);
}
